// MambaDecoderBlock_19155554140859
// MI455X (gfx1250) — compile-verified, matrix operand order checked by reading, not run on hardware
//
#include <hip/hip_runtime.h>
#include <hip/hip_bf16.h>
#include <stdint.h>
#include <stddef.h>

// ---------------------------------------------------------------------------
// Types for CDNA5 WMMA (wave32): v_wmma_f32_16x16x32_bf16
// ---------------------------------------------------------------------------
typedef __bf16 bf16;
typedef __attribute__((ext_vector_type(16))) __bf16 v16bf;
typedef __attribute__((ext_vector_type(8)))  __bf16 v8bf;
typedef __attribute__((ext_vector_type(8)))  float  v8f;

#define DEV __device__ __forceinline__

// Problem dimensions (fixed by the reference)
constexpr int Bq  = 2, Sq = 4096, Dq = 1024;
constexpr int DIq = 2048, DSq = 16, DCq = 4, DTRq = 64;
constexpr int Eq  = 8, Fq = 1024, FSq = 2048;
constexpr int Tq  = Bq * Sq;              // 8192 tokens
constexpr int XDB = DTRq + 2 * DSq;       // 96
constexpr int NCH = 32, CHUNK = Sq / NCH; // chunked scan: 32 x 128
constexpr float EPSq = 1e-6f;

DEV float sigmoidf_(float v) { return 1.f / (1.f + __expf(-v)); }
DEV float siluf_(float v)    { return v * sigmoidf_(v); }
DEV float softplusf_(float v){ return v > 20.f ? v : log1pf(__expf(v)); }

DEV v8f zero8() { v8f z;
#pragma unroll
  for (int i = 0; i < 8; i++) z[i] = 0.f;
  return z; }

#define WMMA_BF16(a, b, c) \
  __builtin_amdgcn_wmma_f32_16x16x32_bf16(false, (a), false, (b), (short)0, (c), false, false)

// ---------------------------------------------------------------------------
// Fragment loaders (ISA 7.12.2 16-bit 16x32 A layout; B loaded from row-major
// W[N,K], i.e. column-major B[K,N], which mirrors the A striping).
// lanes 0-15 : row = lane,   K = k0+0..7  and k0+16..23
// lanes 16-31: row = lane-16,K = k0+8..15 and k0+24..31
// ---------------------------------------------------------------------------
DEV v16bf load_frag_a(const bf16* __restrict__ p0, int ld, int lane) {
  int r  = lane & 15;
  int kh = lane >> 4;
  const bf16* p = p0 + (size_t)r * ld + kh * 8;
  v8bf lo = *reinterpret_cast<const v8bf*>(p);
  v8bf hi = *reinterpret_cast<const v8bf*>(p + 16);
  v16bf f;
#pragma unroll
  for (int i = 0; i < 8; i++) { f[i] = lo[i]; f[i + 8] = hi[i]; }
  return f;
}

template <bool GUARD>
DEV v16bf load_frag_w(const bf16* __restrict__ W, int K, int n0, int k0, int N, int lane) {
  int r  = lane & 15;
  int kh = lane >> 4;
  int n  = n0 + r;
  if (GUARD) {
    v16bf f;
    if (n < N) {
      const bf16* p = W + (size_t)n * K + k0 + kh * 8;
      v8bf lo = *reinterpret_cast<const v8bf*>(p);
      v8bf hi = *reinterpret_cast<const v8bf*>(p + 16);
#pragma unroll
      for (int i = 0; i < 8; i++) { f[i] = lo[i]; f[i + 8] = hi[i]; }
    } else {
#pragma unroll
      for (int i = 0; i < 16; i++) f[i] = (bf16)0.f;
    }
    return f;
  } else {
    const bf16* p = W + (size_t)n * K + k0 + kh * 8;
    v8bf lo = *reinterpret_cast<const v8bf*>(p);
    v8bf hi = *reinterpret_cast<const v8bf*>(p + 16);
    v16bf f;
#pragma unroll
    for (int i = 0; i < 8; i++) { f[i] = lo[i]; f[i + 8] = hi[i]; }
    return f;
  }
}

// ---------------------------------------------------------------------------
// Generic bf16 WMMA GEMM:  C[M,N] = A[M,lda(bf16)] * W[N,K(bf16)]^T
// Block = 256 threads = 8 waves (4 Mwaves x 2 Nwaves); block tile 128 x 128;
// wave tile 32(M) x 64(N) = 2x4 WMMA tiles -> 8 WMMAs per K-step of 32 for
// 6 fragment loads. K-loop is unrolled by 2 with ping-pong fragment buffers
// (K is always a multiple of 64 here), so the software pipeline needs no
// register copies: each stage's loads write the buffer its WMMAs consume.
// Epilogues (compile-time):
//   0: C=v            2: C=v and bf16 copy            3: C=softplus(v+bias[n])
//   4: t=v+aux[idx]; C=t; outf=t (residual dual-store)
//   5: C+=v           6: C+=v*mask[m*E+e]
// ---------------------------------------------------------------------------
template <int EPI, bool GUARD>
__global__ void __launch_bounds__(256)
k_gemm_bf16(const bf16* __restrict__ A, int lda,
            const bf16* __restrict__ W, int K,
            int M, int N, int ldc,
            float* __restrict__ C, const float* __restrict__ aux,
            void* __restrict__ out2, int e_idx) {
  int wave = threadIdx.x >> 5;
  int lane = threadIdx.x & 31;
  int m0 = blockIdx.y * 128 + (wave >> 1) * 32;
  int n0 = blockIdx.x * 128 + (wave & 1) * 64;

  v8f acc[2][4];
#pragma unroll
  for (int i = 0; i < 2; i++)
#pragma unroll
    for (int j = 0; j < 4; j++) acc[i][j] = zero8();

  const bf16* Ar0 = A + (size_t)m0 * lda;
  const bf16* Ar1 = A + (size_t)(m0 + 16) * lda;

  v16bf aE[2], bE[4];   // even K-step (k0)
  v16bf aO[2], bO[4];   // odd  K-step (k0+32)
  aE[0] = load_frag_a(Ar0, lda, lane);
  aE[1] = load_frag_a(Ar1, lda, lane);
#pragma unroll
  for (int j = 0; j < 4; j++) bE[j] = load_frag_w<GUARD>(W, K, n0 + 16 * j, 0, N, lane);

  for (int k0 = 0; k0 < K; k0 += 64) {
    // odd-stage loads (k0+32 always < K since K % 64 == 0)
    aO[0] = load_frag_a(Ar0 + k0 + 32, lda, lane);
    aO[1] = load_frag_a(Ar1 + k0 + 32, lda, lane);
#pragma unroll
    for (int j = 0; j < 4; j++)
      bO[j] = load_frag_w<GUARD>(W, K, n0 + 16 * j, k0 + 32, N, lane);
    __builtin_prefetch(Ar0 + k0 + 64, 0, 1);   // global_prefetch_b8
    __builtin_prefetch(Ar1 + k0 + 64, 0, 1);

    // even-stage math overlaps odd-stage loads
#pragma unroll
    for (int i = 0; i < 2; i++)
#pragma unroll
      for (int j = 0; j < 4; j++) acc[i][j] = WMMA_BF16(aE[i], bE[j], acc[i][j]);

    if (k0 + 64 < K) {  // even-stage loads for next iteration
      aE[0] = load_frag_a(Ar0 + k0 + 64, lda, lane);
      aE[1] = load_frag_a(Ar1 + k0 + 64, lda, lane);
#pragma unroll
      for (int j = 0; j < 4; j++)
        bE[j] = load_frag_w<GUARD>(W, K, n0 + 16 * j, k0 + 64, N, lane);
      __builtin_prefetch(Ar0 + k0 + 96, 0, 1);
      __builtin_prefetch(Ar1 + k0 + 96, 0, 1);
    }

    // odd-stage math overlaps next even-stage loads
#pragma unroll
    for (int i = 0; i < 2; i++)
#pragma unroll
      for (int j = 0; j < 4; j++) acc[i][j] = WMMA_BF16(aO[i], bO[j], acc[i][j]);
  }

  // C/D layout: VGPR r, lanes 0-15 -> M=r, N=lane ; lanes 16-31 -> M=8+r
  int ml = (lane >> 4) << 3;
  int nl = lane & 15;
#pragma unroll
  for (int ti = 0; ti < 2; ti++)
#pragma unroll
    for (int tj = 0; tj < 4; tj++) {
      int nbase = n0 + tj * 16 + nl;
      if (GUARD && nbase >= N) continue;
#pragma unroll
      for (int r = 0; r < 8; r++) {
        int m = m0 + ti * 16 + ml + r;
        float v = acc[ti][tj][r];
        size_t idx = (size_t)m * ldc + nbase;
        if constexpr (EPI == 0) {
          C[idx] = v;
        } else if constexpr (EPI == 2) {
          C[idx] = v; ((bf16*)out2)[idx] = (bf16)v;
        } else if constexpr (EPI == 3) {
          C[idx] = softplusf_(v + aux[nbase]);
        } else if constexpr (EPI == 4) {
          float t = v + aux[idx]; C[idx] = t; ((float*)out2)[idx] = t;
        } else if constexpr (EPI == 5) {
          C[idx] += v;
        } else if constexpr (EPI == 6) {
          C[idx] += v * aux[(size_t)m * Eq + e_idx];
        }
      }
    }
}

// ---------------------------------------------------------------------------
// RMSNorm (f32 in -> bf16 out), one block per token
// ---------------------------------------------------------------------------
__global__ void __launch_bounds__(256)
k_rmsnorm_bf16(const float* __restrict__ x, const float* __restrict__ w,
               bf16* __restrict__ out) {
  int t = blockIdx.x, tid = threadIdx.x;
  __shared__ float red[256];
  const float* row = x + (size_t)t * Dq;
  float ss = 0.f;
  for (int d = tid; d < Dq; d += 256) { float v = row[d]; ss += v * v; }
  red[tid] = ss; __syncthreads();
  for (int s = 128; s > 0; s >>= 1) { if (tid < s) red[tid] += red[tid + s]; __syncthreads(); }
  float scale = rsqrtf(red[0] / (float)Dq + EPSq);
  for (int d = tid; d < Dq; d += 256)
    out[(size_t)t * Dq + d] = (bf16)(row[d] * scale * w[d]);
}

// ---------------------------------------------------------------------------
// Causal depthwise conv (DC=4) + SiLU, reads xi half of xz [T, 2*DI]
// ---------------------------------------------------------------------------
__global__ void __launch_bounds__(256)
k_conv_silu(const float* __restrict__ xz, const float* __restrict__ cw,
            const float* __restrict__ cb, float* __restrict__ xif,
            bf16* __restrict__ xib) {
  size_t i = (size_t)blockIdx.x * 256 + threadIdx.x;
  if (i >= (size_t)Bq * Sq * DIq) return;
  int d = (int)(i % DIq);
  size_t bs = i / DIq;
  int s = (int)(bs % Sq);
  int b = (int)(bs / Sq);
  const float* col = xz + (size_t)b * Sq * (2 * DIq) + d;
  float acc = cb[d];
#pragma unroll
  for (int j = 0; j < DCq; j++) {
    int ss = s - (DCq - 1) + j;
    if (ss >= 0) acc += cw[d * DCq + j] * col[(size_t)ss * (2 * DIq)];
  }
  float v = siluf_(acc);
  xif[i] = v;
  xib[i] = (bf16)v;
}

// ---------------------------------------------------------------------------
// Chunked selective scan (3 passes). State DS=16 per (b,d) in registers.
// dA = exp(dt*A) and dBu = dt*B*xi are recomputed on the fly (never stored:
// materializing them would be 2 x 2.1 GB of HBM traffic).
// ---------------------------------------------------------------------------
__global__ void __launch_bounds__(64)
k_scan1(const float* __restrict__ dt, const float* __restrict__ xdbl,
        const float* __restrict__ xi, const float* __restrict__ A_log,
        float* __restrict__ hch, float* __restrict__ apr) {
  int chunk = blockIdx.x;
  int d = blockIdx.y * 64 + threadIdx.x;
  int b = blockIdx.z;
  float A[DSq], h[DSq], ap[DSq];
#pragma unroll
  for (int n = 0; n < DSq; n++) {
    A[n] = -__expf(A_log[(size_t)d * DSq + n]); h[n] = 0.f; ap[n] = 1.f;
  }
  int s0 = chunk * CHUNK;
  for (int s = s0; s < s0 + CHUNK; s++) {
    size_t tt = (size_t)b * Sq + s;
    float dtv = dt[tt * DIq + d];
    float xiv = xi[tt * DIq + d];
    const float* Brow = xdbl + tt * XDB + DTRq;
    float dx = dtv * xiv;
#pragma unroll
    for (int n = 0; n < DSq; n++) {
      float a = __expf(dtv * A[n]);
      h[n] = a * h[n] + dx * Brow[n];
      ap[n] *= a;
    }
  }
  size_t base = (((size_t)b * NCH + chunk) * DIq + d) * DSq;
#pragma unroll
  for (int n = 0; n < DSq; n++) { hch[base + n] = h[n]; apr[base + n] = ap[n]; }
}

__global__ void __launch_bounds__(256)
k_scan2(const float* __restrict__ hch, const float* __restrict__ apr,
        float* __restrict__ hst) {
  int idx = blockIdx.x * 256 + threadIdx.x;   // b*DI + d
  int b = idx / DIq, d = idx % DIq;
  float hs[DSq];
#pragma unroll
  for (int n = 0; n < DSq; n++) hs[n] = 0.f;
  for (int c = 0; c < NCH; c++) {
    size_t base = (((size_t)b * NCH + c) * DIq + d) * DSq;
#pragma unroll
    for (int n = 0; n < DSq; n++) {
      hst[base + n] = hs[n];
      hs[n] = apr[base + n] * hs[n] + hch[base + n];
    }
  }
}

__global__ void __launch_bounds__(64)
k_scan3(const float* __restrict__ dt, const float* __restrict__ xdbl,
        const float* __restrict__ xi, const float* __restrict__ xz,
        const float* __restrict__ A_log, const float* __restrict__ Dskip,
        const float* __restrict__ hst, bf16* __restrict__ yout) {
  int chunk = blockIdx.x;
  int d = blockIdx.y * 64 + threadIdx.x;
  int b = blockIdx.z;
  float A[DSq], h[DSq];
  size_t base = (((size_t)b * NCH + chunk) * DIq + d) * DSq;
#pragma unroll
  for (int n = 0; n < DSq; n++) {
    A[n] = -__expf(A_log[(size_t)d * DSq + n]);
    h[n] = hst[base + n];
  }
  float dsk = Dskip[d];
  int s0 = chunk * CHUNK;
  for (int s = s0; s < s0 + CHUNK; s++) {
    size_t tt = (size_t)b * Sq + s;
    float dtv = dt[tt * DIq + d];
    float xiv = xi[tt * DIq + d];
    const float* Brow = xdbl + tt * XDB + DTRq;
    const float* Crow = Brow + DSq;
    float dx = dtv * xiv, y = 0.f;
#pragma unroll
    for (int n = 0; n < DSq; n++) {
      float a = __expf(dtv * A[n]);
      h[n] = a * h[n] + dx * Brow[n];
      y += h[n] * Crow[n];
    }
    float res = xz[tt * (2 * DIq) + DIq + d];
    float o = (y + xiv * dsk) * siluf_(res);   // fused D-skip + gate
    yout[tt * DIq + d] = (bf16)o;
  }
}

// ---------------------------------------------------------------------------
// Router: softmax over 8 experts, top-2 by (softmax+bias), weights from
// softmax renormalized -> dense mask[T,8] (0 except the two picked experts)
// ---------------------------------------------------------------------------
__global__ void __launch_bounds__(256)
k_gate(const bf16* __restrict__ n2, const float* __restrict__ gw,
       const float* __restrict__ gb, float* __restrict__ mask) {
  int t = blockIdx.x, tid = threadIdx.x;
  __shared__ float red[256];
  __shared__ float sc[Eq];
  float p[Eq];
#pragma unroll
  for (int e = 0; e < Eq; e++) p[e] = 0.f;
  for (int d = tid; d < Dq; d += 256) {
    float xv = (float)n2[(size_t)t * Dq + d];
#pragma unroll
    for (int e = 0; e < Eq; e++) p[e] += xv * gw[(size_t)e * Dq + d];
  }
  for (int e = 0; e < Eq; e++) {
    red[tid] = p[e]; __syncthreads();
    for (int s = 128; s > 0; s >>= 1) { if (tid < s) red[tid] += red[tid + s]; __syncthreads(); }
    if (tid == 0) sc[e] = red[0];
    __syncthreads();
  }
  if (tid == 0) {
    float mx = sc[0];
    for (int e = 1; e < Eq; e++) mx = fmaxf(mx, sc[e]);
    float sum = 0.f, pr[Eq];
    for (int e = 0; e < Eq; e++) { pr[e] = __expf(sc[e] - mx); sum += pr[e]; }
    for (int e = 0; e < Eq; e++) pr[e] /= sum;
    int i0 = 0; float k0 = -1e30f;
    for (int e = 0; e < Eq; e++) { float kk = pr[e] + gb[e]; if (kk > k0) { k0 = kk; i0 = e; } }
    int i1 = 0; float k1 = -1e30f;
    for (int e = 0; e < Eq; e++) if (e != i0) { float kk = pr[e] + gb[e]; if (kk > k1) { k1 = kk; i1 = e; } }
    float w0 = pr[i0], w1 = pr[i1], wsum = w0 + w1 + 1e-9f;
    for (int e = 0; e < Eq; e++) mask[(size_t)t * Eq + e] = 0.f;
    mask[(size_t)t * Eq + i0] = w0 / wsum;
    mask[(size_t)t * Eq + i1] = w1 / wsum;
  }
}

// ---------------------------------------------------------------------------
// Elementwise helpers
// ---------------------------------------------------------------------------
__global__ void __launch_bounds__(256)
k_f32_to_bf16(const float* __restrict__ in, bf16* __restrict__ out, size_t n) {
  size_t i = (size_t)blockIdx.x * 256 + threadIdx.x;
  if (i < n) out[i] = (bf16)in[i];
}

__global__ void __launch_bounds__(256)
k_silu_mul_bf16(const float* __restrict__ g, const float* __restrict__ u,
                bf16* __restrict__ h, size_t n) {
  size_t i = (size_t)blockIdx.x * 256 + threadIdx.x;
  if (i < n) h[i] = (bf16)(siluf_(g[i]) * u[i]);
}

// ---------------------------------------------------------------------------
// Launch
// ---------------------------------------------------------------------------
extern "C" void kernel_launch(void* const* d_in, const int* in_sizes, int n_in,
                              void* d_out, int out_size, void* d_ws, size_t ws_size,
                              hipStream_t stream) {
  const float* x         = (const float*)d_in[0];
  const float* norm_w    = (const float*)d_in[1];
  const float* in_proj_w = (const float*)d_in[2];
  const float* conv_w    = (const float*)d_in[3];
  const float* conv_b    = (const float*)d_in[4];
  const float* x_proj_w  = (const float*)d_in[5];
  const float* dt_proj_w = (const float*)d_in[6];
  const float* dt_proj_b = (const float*)d_in[7];
  const float* A_log     = (const float*)d_in[8];
  const float* D_skip    = (const float*)d_in[9];
  const float* out_proj_w= (const float*)d_in[10];
  const float* gate_w    = (const float*)d_in[11];
  const float* gate_bias = (const float*)d_in[12];
  const float* w_gate    = (const float*)d_in[13];
  const float* w_up      = (const float*)d_in[14];
  const float* w_down    = (const float*)d_in[15];
  const float* ws_gate   = (const float*)d_in[16];
  const float* ws_up     = (const float*)d_in[17];
  const float* ws_down   = (const float*)d_in[18];
  float* outp = (float*)d_out;

  char* base = (char*)d_ws; size_t off = 0;
  auto alloc = [&](size_t bytes) -> char* {
    char* p = base + off;
    off = (off + bytes + 255) & ~(size_t)255;
    return p;
  };

  // bf16 weight copies (~73MiB total -> resident in 192MB L2)
  bf16* wb_in  = (bf16*)alloc((size_t)2 * DIq * Dq * 2);
  bf16* wb_xp  = (bf16*)alloc((size_t)XDB * DIq * 2);
  bf16* wb_dt  = (bf16*)alloc((size_t)DIq * DTRq * 2);
  bf16* wb_out = (bf16*)alloc((size_t)Dq * DIq * 2);
  bf16* wb_g   = (bf16*)alloc((size_t)Eq * Fq * Dq * 2);
  bf16* wb_u   = (bf16*)alloc((size_t)Eq * Fq * Dq * 2);
  bf16* wb_d   = (bf16*)alloc((size_t)Eq * Dq * Fq * 2);
  bf16* wb_sg  = (bf16*)alloc((size_t)FSq * Dq * 2);
  bf16* wb_su  = (bf16*)alloc((size_t)FSq * Dq * 2);
  bf16* wb_sd  = (bf16*)alloc((size_t)Dq * FSq * 2);
  // activations
  bf16*  nrm_bf  = (bf16*)alloc((size_t)Tq * Dq * 2);
  float* xz      = (float*)alloc((size_t)Tq * 2 * DIq * 4);
  float* xi_f    = (float*)alloc((size_t)Tq * DIq * 4);
  bf16*  xi_b    = (bf16*)alloc((size_t)Tq * DIq * 2);
  float* xdbl    = (float*)alloc((size_t)Tq * XDB * 4);
  bf16*  xdbl_b  = (bf16*)alloc((size_t)Tq * XDB * 2);
  float* dtbuf   = (float*)alloc((size_t)Tq * DIq * 4);
  float* hchunk  = (float*)alloc((size_t)Bq * NCH * DIq * DSq * 4);
  float* aprod   = (float*)alloc((size_t)Bq * NCH * DIq * DSq * 4);
  float* hstart  = (float*)alloc((size_t)Bq * NCH * DIq * DSq * 4);
  bf16*  yg_b    = (bf16*)alloc((size_t)Tq * DIq * 2);
  float* m_ws    = (float*)alloc((size_t)Tq * Dq * 4);
  bf16*  n2_bf   = (bf16*)alloc((size_t)Tq * Dq * 2);
  float* maskb   = (float*)alloc((size_t)Tq * Eq * 4);
  float* gbuf    = (float*)alloc((size_t)Tq * FSq * 4);
  float* ubuf    = (float*)alloc((size_t)Tq * FSq * 4);
  bf16*  h_bf    = (bf16*)alloc((size_t)Tq * FSq * 2);

  auto cvt = [&](const float* s, bf16* d, size_t n) {
    k_f32_to_bf16<<<dim3((unsigned)((n + 255) / 256)), 256, 0, stream>>>(s, d, n);
  };
  auto grid = [](int M, int N) { return dim3((N + 127) / 128, M / 128); };

  // ---- weight quantization (deterministic each call) ----
  cvt(in_proj_w,  wb_in,  (size_t)2 * DIq * Dq);
  cvt(x_proj_w,   wb_xp,  (size_t)XDB * DIq);
  cvt(dt_proj_w,  wb_dt,  (size_t)DIq * DTRq);
  cvt(out_proj_w, wb_out, (size_t)Dq * DIq);
  cvt(w_gate,     wb_g,   (size_t)Eq * Fq * Dq);
  cvt(w_up,       wb_u,   (size_t)Eq * Fq * Dq);
  cvt(w_down,     wb_d,   (size_t)Eq * Dq * Fq);
  cvt(ws_gate,    wb_sg,  (size_t)FSq * Dq);
  cvt(ws_up,      wb_su,  (size_t)FSq * Dq);
  cvt(ws_down,    wb_sd,  (size_t)Dq * FSq);

  // ---- Mamba ----
  k_rmsnorm_bf16<<<Tq, 256, 0, stream>>>(x, norm_w, nrm_bf);
  k_gemm_bf16<0, false><<<grid(Tq, 2 * DIq), 256, 0, stream>>>(
      nrm_bf, Dq, wb_in, Dq, Tq, 2 * DIq, 2 * DIq, xz, nullptr, nullptr, 0);
  {
    size_t n = (size_t)Tq * DIq;
    k_conv_silu<<<dim3((unsigned)((n + 255) / 256)), 256, 0, stream>>>(
        xz, conv_w, conv_b, xi_f, xi_b);
  }
  // x_proj: N=96 needs the guarded path (only GEMM that does)
  k_gemm_bf16<2, true><<<grid(Tq, XDB), 256, 0, stream>>>(
      xi_b, DIq, wb_xp, DIq, Tq, XDB, XDB, xdbl, nullptr, (void*)xdbl_b, 0);
  k_gemm_bf16<3, false><<<grid(Tq, DIq), 256, 0, stream>>>(
      xdbl_b, XDB, wb_dt, DTRq, Tq, DIq, DIq, dtbuf, dt_proj_b, nullptr, 0);

  k_scan1<<<dim3(NCH, DIq / 64, Bq), 64, 0, stream>>>(dtbuf, xdbl, xi_f, A_log, hchunk, aprod);
  k_scan2<<<dim3(Bq * DIq / 256), 256, 0, stream>>>(hchunk, aprod, hstart);
  k_scan3<<<dim3(NCH, DIq / 64, Bq), 64, 0, stream>>>(dtbuf, xdbl, xi_f, xz, A_log,
                                                      D_skip, hstart, yg_b);
  // out_proj + residual: writes m (for norm2) and initializes d_out = m
  k_gemm_bf16<4, false><<<grid(Tq, Dq), 256, 0, stream>>>(
      yg_b, DIq, wb_out, DIq, Tq, Dq, Dq, m_ws, x, (void*)outp, 0);

  // ---- MoE ----
  k_rmsnorm_bf16<<<Tq, 256, 0, stream>>>(m_ws, norm_w, n2_bf);
  k_gate<<<Tq, 256, 0, stream>>>(n2_bf, gate_w, gate_bias, maskb);

  // shared expert
  k_gemm_bf16<0, false><<<grid(Tq, FSq), 256, 0, stream>>>(
      n2_bf, Dq, wb_sg, Dq, Tq, FSq, FSq, gbuf, nullptr, nullptr, 0);
  k_gemm_bf16<0, false><<<grid(Tq, FSq), 256, 0, stream>>>(
      n2_bf, Dq, wb_su, Dq, Tq, FSq, FSq, ubuf, nullptr, nullptr, 0);
  {
    size_t n = (size_t)Tq * FSq;
    k_silu_mul_bf16<<<dim3((unsigned)((n + 255) / 256)), 256, 0, stream>>>(gbuf, ubuf, h_bf, n);
  }
  k_gemm_bf16<5, false><<<grid(Tq, Dq), 256, 0, stream>>>(
      h_bf, FSq, wb_sd, FSq, Tq, Dq, Dq, outp, nullptr, nullptr, 0);

  // routed experts (dense-masked; epilogue scales by mask[t,e] and accumulates)
  for (int e = 0; e < Eq; e++) {
    const bf16* wg = wb_g + (size_t)e * Fq * Dq;
    const bf16* wu = wb_u + (size_t)e * Fq * Dq;
    const bf16* wd = wb_d + (size_t)e * Dq * Fq;
    k_gemm_bf16<0, false><<<grid(Tq, Fq), 256, 0, stream>>>(
        n2_bf, Dq, wg, Dq, Tq, Fq, Fq, gbuf, nullptr, nullptr, 0);
    k_gemm_bf16<0, false><<<grid(Tq, Fq), 256, 0, stream>>>(
        n2_bf, Dq, wu, Dq, Tq, Fq, Fq, ubuf, nullptr, nullptr, 0);
    size_t n = (size_t)Tq * Fq;
    k_silu_mul_bf16<<<dim3((unsigned)((n + 255) / 256)), 256, 0, stream>>>(gbuf, ubuf, h_bf, n);
    k_gemm_bf16<6, false><<<grid(Tq, Dq), 256, 0, stream>>>(
        h_bf, Fq, wd, Fq, Tq, Dq, Dq, outp, maskb, nullptr, e);
  }
}
